// Conv2dPCCNorm_29291676959208
// MI455X (gfx1250) — matrix operand order repaired and checked by reading, not compiled, not run on hardware
//
#include <hip/hip_runtime.h>
#include <math.h>

typedef __bf16 v16bf __attribute__((ext_vector_type(16)));
typedef __bf16 v8bf  __attribute__((ext_vector_type(8)));
typedef __bf16 v4bf  __attribute__((ext_vector_type(4)));
typedef float  v8f   __attribute__((ext_vector_type(8)));

#define BB   32          // batch
#define CC   128         // in channels
#define HH   56
#define WW   56
#define OO   256         // out channels
#define HWSZ (HH*WW)     // 3136
#define KK   (CC*9)      // 1152
#define NPIX (BB*HWSZ)   // 100352

// ---------------- kernel 1: per-sample mean of x ----------------
__global__ void xmean_kernel(const float* __restrict__ x, float* __restrict__ mean_x) {
    __shared__ float red[256];
    const int b = blockIdx.x;
    const int t = threadIdx.x;
    const float* xb = x + (size_t)b * (CC * HWSZ);
    float s = 0.f;
    for (int i = t; i < CC * HWSZ; i += 256) s += xb[i];
    red[t] = s;
    __syncthreads();
    for (int k = 128; k > 0; k >>= 1) {
        if (t < k) red[t] += red[t + k];
        __syncthreads();
    }
    if (t == 0) mean_x[b] = red[0] / (float)(CC * HWSZ);
}

// ---------------- kernel 2: weight prep -------------------------
// wc_t: bf16 centered weights, layout [O][r=kh*3+kw][c] (K-slices contiguous in c);
// wsum: [O][3][3] border-masked ||w_c||^2
__global__ void wprep_kernel(const float* __restrict__ w, __bf16* __restrict__ wc_t,
                             float* __restrict__ wsum) {
    __shared__ float red[128];
    __shared__ float s9[9];
    const int o = blockIdx.x;
    const int t = threadIdx.x;   // 128 threads
    const float* wo = w + (size_t)o * KK;

    if (t < 9) s9[t] = 0.f;
    float s = 0.f;
    for (int k = t; k < KK; k += 128) s += wo[k];
    red[t] = s;
    __syncthreads();
    for (int k = 64; k > 0; k >>= 1) {
        if (t < k) red[t] += red[t + k];
        __syncthreads();
    }
    const float mean = red[0] / (float)KK;

    for (int k = t; k < KK; k += 128) {
        float v = wo[k] - mean;                       // k = c*9 + r
        wc_t[(size_t)o * KK + (k % 9) * CC + (k / 9)] = (__bf16)v;
        atomicAdd(&s9[k % 9], v * v);                 // ds_add_f32
    }
    __syncthreads();

    if (t < 9) {
        int hi = t / 3, wi = t % 3;
        float acc = 0.f;
        for (int kh = 0; kh < 3; ++kh) {
            bool hok = (hi == 1) || (hi == 0 && kh > 0) || (hi == 2 && kh < 2);
            if (!hok) continue;
            for (int kw = 0; kw < 3; ++kw) {
                bool wok = (wi == 1) || (wi == 0 && kw > 0) || (wi == 2 && kw < 2);
                if (wok) acc += s9[kh * 3 + kw];
            }
        }
        wsum[o * 9 + t] = acc;
    }
}

// ---------------- kernel 3: centered bf16 x (NHWC) + per-pixel sum of squares ---
__global__ void xprep_kernel(const float* __restrict__ x, const float* __restrict__ mean_x,
                             __bf16* __restrict__ xc, float* __restrict__ sq) {
    const int pn = blockIdx.x * 256 + threadIdx.x;   // 0 .. NPIX-1
    const int b  = pn / HWSZ;
    const int hw = pn % HWSZ;
    const float mean = mean_x[b];
    float acc = 0.f;
    const size_t ibase = (size_t)b * CC * HWSZ + hw;
    __bf16* orow = xc + (size_t)pn * CC;
    for (int c8 = 0; c8 < CC; c8 += 8) {
        v8bf v;
#pragma unroll
        for (int j = 0; j < 8; ++j) {
            float f = x[ibase + (size_t)(c8 + j) * HWSZ] - mean;
            __bf16 bv = (__bf16)f;
            v[j] = bv;
            float fv = (float)bv;
            acc += fv * fv;
        }
        *(v8bf*)(orow + c8) = v;                     // contiguous 16B store
    }
    sq[pn] = acc;
}

// ---------------- kernel 4: 3x3 box sum of sq -> x2_len ------------------------
__global__ void boxsum_kernel(const float* __restrict__ sq, float* __restrict__ x2n) {
    const int pn = blockIdx.x * 256 + threadIdx.x;
    const int b = pn / HWSZ;
    const int hw = pn % HWSZ;
    const int h = hw / WW, w = hw % WW;
    float acc = 0.f;
    for (int dh = -1; dh <= 1; ++dh) {
        int ih = h + dh;
        if (ih < 0 || ih >= HH) continue;
        for (int dw = -1; dw <= 1; ++dw) {
            int iw = w + dw;
            if (iw < 0 || iw >= WW) continue;
            acc += sq[b * HWSZ + ih * WW + iw];
        }
    }
    x2n[pn] = acc;
}

// ---------------- kernel 5: main implicit-GEMM conv via bf16 WMMA --------------
// Block tile: 32 out-channels (M) x 128 pixels (N). 8 waves; each wave owns one
// 16-row M sub-tile and two 16-col N sub-tiles (A-fragment reuse -> 2 WMMAs/step).
// K = 9 slabs (kh,kw) x 4 chunks of 32 channels = 36 steps.
// Software pipeline, 2 deep: g-load(s+2) | lds-store(s+1) | compute(s).
__global__ __launch_bounds__(256)
void pcconv_kernel(const __bf16* __restrict__ xc, const __bf16* __restrict__ wct,
                   const float* __restrict__ x2n, const float* __restrict__ wsum,
                   float* __restrict__ out) {
    __shared__ __bf16 lw[2][32 * 40];    // rows padded to 40 hw (80B) -> aligned b128
    __shared__ __bf16 lx[2][128 * 40];

    const int t    = threadIdx.x;
    const int lane = t & 31;
    const int wv   = t >> 5;
    const int wm   = wv >> 2;            // 0..1  channel sub-tile
    const int wn   = wv & 3;             // 0..3  -> pixels [wn*32, wn*32+32)
    const int lm   = lane & 15;
    const int half = lane >> 4;

    const int n_base = blockIdx.x * 128;
    const int o_base = blockIdx.y * 32;

    // ---- per-thread staging geometry (computed ONCE) ----
    const int px  = t & 127;
    const int cg  = t >> 7;              // 0 or 1
    const int pn  = n_base + px;
    const int phw = pn % HWSZ;
    const int ph  = phw / WW;
    const int pw  = phw % WW;
    const int code = (ph == 0 ? 1 : 0) | (ph == HH - 1 ? 2 : 0)
                   | (pw == 0 ? 4 : 0) | (pw == WW - 1 ? 8 : 0);
    const long long ppix = (long long)pn * CC + cg * 16;   // NHWC row + channel group
    __bf16* const lx_w = &lx[0][px * 40 + cg * 16];
    const int ol  = t >> 3;
    const int kks = (t & 7) * 4;
    const size_t wrow = (size_t)(o_base + ol) * KK + kks;
    __bf16* const lw_w = &lw[0][ol * 40 + kks];

    // fragment LDS offsets (halfwords, constant per thread)
    const int a_off  = (wm * 16 + lm) * 40 + half * 16;  // + {0,8}
    const int b0_off = (wn * 32 + lm) * 40 + half * 16;  // + {0,8}
    const int b1_off = b0_off + 16 * 40;

    v8f acc0 = {}, acc1 = {};

    // staged-tile registers (live across one compute step)
    v8bf sx0 = {}, sx1 = {};
    v4bf sw4;

    auto gload = [&](int s) {
        const int slab = s >> 2;                 // 0..8  (kh*3+kw), SALU only
        const int c0   = (s & 3) * 32;
        const int kh   = slab / 3, kw = slab % 3;
        const int spat = (kh - 1) * WW + (kw - 1);       // pixel shift
        const int smask = (kh == 0 ? 1 : 0) | (kh == 2 ? 2 : 0)
                        | (kw == 0 ? 4 : 0) | (kw == 2 ? 8 : 0);
        sw4 = *(const v4bf*)(wct + wrow + slab * CC + c0);
        v8bf z = {};
        sx0 = z; sx1 = z;
        if ((code & smask) == 0) {
            const __bf16* p = xc + (ppix + (long long)spat * CC + c0);
            sx0 = *(const v8bf*)p;
            sx1 = *(const v8bf*)(p + 8);
        }
    };
    auto sstore = [&](int s) {
        const int buf = s & 1;
        *(v4bf*)(lw_w + buf * (32 * 40)) = sw4;
        __bf16* dst = lx_w + buf * (128 * 40);
        *(v8bf*)dst       = sx0;
        *(v8bf*)(dst + 8) = sx1;
    };

    gload(0);
    sstore(0);
    gload(1);

    for (int s = 0; s < 36; ++s) {
        __syncthreads();                 // buf[s&1] ready; buf[(s+1)&1] free

        const __bf16* lwb = &lw[s & 1][0];
        const __bf16* lxb = &lx[s & 1][0];
        // A fragment (16-bit A 16x32 layout)
        v8bf a0 = *(const v8bf*)(lwb + a_off);
        v8bf a1 = *(const v8bf*)(lwb + a_off + 8);
        v16bf av = __builtin_shufflevector(a0, a1, 0,1,2,3,4,5,6,7,8,9,10,11,12,13,14,15);
        // B fragments (16-bit B 32x16 layout), two pixel sub-tiles
        v8bf b00 = *(const v8bf*)(lxb + b0_off);
        v8bf b01 = *(const v8bf*)(lxb + b0_off + 8);
        v16bf bv0 = __builtin_shufflevector(b00, b01, 0,1,2,3,4,5,6,7,8,9,10,11,12,13,14,15);
        v8bf b10 = *(const v8bf*)(lxb + b1_off);
        v8bf b11 = *(const v8bf*)(lxb + b1_off + 8);
        v16bf bv1 = __builtin_shufflevector(b10, b11, 0,1,2,3,4,5,6,7,8,9,10,11,12,13,14,15);

        acc0 = __builtin_amdgcn_wmma_f32_16x16x32_bf16(
                   false, av, false, bv0, (short)0, acc0, false, false);
        acc1 = __builtin_amdgcn_wmma_f32_16x16x32_bf16(
                   false, av, false, bv1, (short)0, acc1, false, false);

        // stash tile s+1 into the other buffer, then start fetching tile s+2
        if (s + 1 < 36) sstore(s + 1);
        if (s + 2 < 36) gload(s + 2);
    }

    // --- epilogue: normalize, clamp, coalesced NCHW stores ---
#pragma unroll
    for (int sub = 0; sub < 2; ++sub) {
        const v8f acc = sub ? acc1 : acc0;
        const int qn = n_base + wn * 32 + sub * 16 + lm;
        const int b  = qn / HWSZ;
        const int hw = qn % HWSZ;
        const int h  = hw / WW, w = hw % WW;
        const int hi = (h == 0) ? 0 : ((h == HH - 1) ? 2 : 1);
        const int wi = (w == 0) ? 0 : ((w == WW - 1) ? 2 : 1);
        const float rx = rsqrtf(x2n[qn]);
#pragma unroll
        for (int r = 0; r < 8; ++r) {
            int o = o_base + wm * 16 + r + 8 * half;
            float w2 = wsum[o * 9 + hi * 3 + wi];
            float val = acc[r] * rx * rsqrtf(w2);
            val = fmaxf(val, 1e-10f);
            out[((size_t)(b * OO + o)) * HWSZ + hw] = val;
        }
    }
}

// ---------------- launch -------------------------------------------------------
extern "C" void kernel_launch(void* const* d_in, const int* in_sizes, int n_in,
                              void* d_out, int out_size, void* d_ws, size_t ws_size,
                              hipStream_t stream) {
    const float* x = (const float*)d_in[0];
    const float* w = (const float*)d_in[1];
    float* out = (float*)d_out;

    char* ws = (char*)d_ws;
    size_t off = 0;
    float* mean_x = (float*)(ws + off);            off += 256;                         // 32 f
    __bf16* wc_t  = (__bf16*)(ws + off);           off += (size_t)OO * KK * 2;         // 576KB
    float* wsum   = (float*)(ws + off);            off += (size_t)OO * 9 * 4;          // 9KB
    off = (off + 4095) & ~(size_t)4095;
    __bf16* xcbf  = (__bf16*)(ws + off);           off += (size_t)NPIX * CC * 2;       // 24.5MB (NHWC)
    float* sq     = (float*)(ws + off);            off += (size_t)NPIX * 4;            // 392KB
    float* x2n    = (float*)(ws + off);            off += (size_t)NPIX * 4;            // 392KB
    (void)ws_size; (void)in_sizes; (void)n_in; (void)out_size;

    xmean_kernel <<<BB, 256, 0, stream>>>(x, mean_x);
    wprep_kernel <<<OO, 128, 0, stream>>>(w, wc_t, wsum);
    xprep_kernel <<<NPIX / 256, 256, 0, stream>>>(x, mean_x, xcbf, sq);
    boxsum_kernel<<<NPIX / 256, 256, 0, stream>>>(sq, x2n);

    dim3 grid(NPIX / 128, OO / 32);
    pcconv_kernel<<<grid, 256, 0, stream>>>(xcbf, wc_t, x2n, wsum, out);
}
